// SpectralConvS2_3229815406938
// MI455X (gfx1250) — matrix operand order repaired and checked by reading, not compiled, not run on hardware
//
#include <hip/hip_runtime.h>
#include <hip/hip_bf16.h>

// ---------------------------------------------------------------------------
// SpectralConvS2 + LoRA for MI455X (gfx1250).
// Pipeline:
//   A) row rFFT-512 (half-size complex trick)            x -> Xrow
//   B) col FFT-256                                        Xrow -> Xs[x][i][n]
//   W) W_eff[x][o][i] = W[i][o][x] + sum_r A[i][r][x]B[r][o][x]
//   C) per-mode complex GEMM, bf16x3 WMMA, TDM-staged     Ys[x][o][n]
//   D) col iFFT-256 (1/256)                               Ys -> Yrow
//   E) row irFFT-512 (inverse pack, 1/256)                Yrow -> out
//   R) residual = x (D2D copy)
// ---------------------------------------------------------------------------

#define DEVFN __device__ __forceinline__

#if defined(__gfx1250__) && __has_builtin(__builtin_amdgcn_tensor_load_to_lds)
#define USE_TDM 1
#else
#define USE_TDM 0
#endif

namespace sc {

constexpr int Bsz  = 2;
constexpr int CH   = 256;   // channels
constexpr int RANK = 16;
constexpr int NLAT = 256;
constexpr int NLON = 512;
constexpr int WC   = NLON / 2 + 1;     // 257
constexpr int NCOL = Bsz * WC;         // 514 (b,y) columns
constexpr int NP   = 544;              // padded N (17 * 32)
constexpr float PI = 3.14159265358979323846f;

struct alignas(8) cplx { float x, y; };

DEVFN cplx cmk(float a, float b) { cplx c; c.x = a; c.y = b; return c; }
DEVFN cplx cmadd(cplx a, cplx b, cplx acc) {
  acc.x += a.x * b.x - a.y * b.y;
  acc.y += a.x * b.y + a.y * b.x;
  return acc;
}
DEVFN cplx ldc2(const float* p, size_t cidx) {
  return cmk(p[2 * cidx], p[2 * cidx + 1]);
}

// ------------------------------ shared FFT-256 ------------------------------
DEVFN void fft256(cplx* s, float sign) {
  int t = threadIdx.x;
#pragma unroll
  for (int st = 0; st < 8; ++st) {
    int mh   = 1 << st;
    int pos  = t & (mh - 1);
    int base = ((t >> st) << (st + 1)) + pos;
    float ang = sign * PI * (float)pos / (float)mh;
    float cs, sn;
    __sincosf(ang, &cs, &sn);
    cplx a = s[base], b = s[base + mh];
    cplx wb = cmk(b.x * cs - b.y * sn, b.x * sn + b.y * cs);
    s[base]      = cmk(a.x + wb.x, a.y + wb.y);
    s[base + mh] = cmk(a.x - wb.x, a.y - wb.y);
    __syncthreads();
  }
}

DEVFN int brev8(int n) { return (int)(__brev((unsigned)n) >> 24); }

// ------------------------------ Stage A: row rFFT ---------------------------
__global__ __launch_bounds__(128) void fft_rows_fwd(const float* __restrict__ x,
                                                    cplx* __restrict__ Xrow) {
  __shared__ cplx s[256];
  size_t row = blockIdx.x;                       // (b*CH + c)*NLAT + lat
  const float* xr = x + row * NLON;
  int t = threadIdx.x;
  for (int n = t; n < 256; n += 128)
    s[brev8(n)] = cmk(xr[2 * n], xr[2 * n + 1]);
  __syncthreads();
  fft256(s, -1.0f);
  cplx* out = Xrow + row * WC;
  for (int k = t; k < WC; k += 128) {
    int km  = (256 - k) & 255;
    cplx Zk = s[k & 255];
    cplx Zm = s[km];
    cplx E = cmk(0.5f * (Zk.x + Zm.x), 0.5f * (Zk.y - Zm.y));
    cplx O = cmk(0.5f * (Zk.x - Zm.x), 0.5f * (Zk.y + Zm.y));
    float cs, sn;
    __sincosf(-PI * (float)k / 256.0f, &cs, &sn);
    float p = O.x * cs - O.y * sn;
    float q = O.x * sn + O.y * cs;
    out[k] = cmk(E.x + q, E.y - p);              // E + (-i)(p+iq)
  }
}

// ------------------------------ Stage B: col FFT ----------------------------
__global__ __launch_bounds__(128) void fft_cols_fwd(const cplx* __restrict__ Xrow,
                                                    cplx* __restrict__ Xs) {
  __shared__ cplx s[256];
  int idx = blockIdx.x;                          // (b*CH + c)*WC + y
  int y = idx % WC, bc = idx / WC;
  int c = bc % CH, b = bc / CH;
  const cplx* colp = Xrow + (size_t)bc * NLAT * WC + y;
  int t = threadIdx.x;
  for (int n = t; n < 256; n += 128) s[brev8(n)] = colp[(size_t)n * WC];
  __syncthreads();
  fft256(s, -1.0f);
  int ncol = b * WC + y;
  cplx* outp = Xs + ncol;
  for (int k = t; k < 256; k += 128)
    outp[((size_t)k * CH + c) * NP] = s[k];
}

// ------------------------- Stage W: fold LoRA into W ------------------------
__global__ __launch_bounds__(256) void weff_kernel(const float* __restrict__ W,
                                                   const float* __restrict__ A,
                                                   const float* __restrict__ Bl,
                                                   cplx* __restrict__ Weff) {
  __shared__ cplx Ash[CH][RANK];                 // 32 KB
  int x = blockIdx.x;
  int t = threadIdx.x;                           // t == o
  for (int e = t; e < CH * RANK; e += 256) {
    int i = e >> 4, r = e & 15;
    Ash[i][r] = ldc2(A, ((size_t)(i * RANK + r)) * NLAT + x);
  }
  cplx breg[RANK];
#pragma unroll
  for (int r = 0; r < RANK; ++r)
    breg[r] = ldc2(Bl, ((size_t)(r * CH + t)) * NLAT + x);
  __syncthreads();
  cplx* out = Weff + ((size_t)x * CH + t) * CH;
  for (int i = 0; i < CH; ++i) {
    cplx acc = ldc2(W, ((size_t)(i * CH + t)) * NLAT + x);
#pragma unroll
    for (int r = 0; r < RANK; ++r) acc = cmadd(Ash[i][r], breg[r], acc);
    out[i] = acc;
  }
}

// ----------------------- Stage pad: zero Xs pad columns ---------------------
__global__ void pad_zero(cplx* __restrict__ Xs) {
  constexpr int PADW = NP - NCOL;                // 30
  int i = blockIdx.x * blockDim.x + threadIdx.x;
  if (i >= NLAT * CH * PADW) return;
  int p = i % PADW, xc = i / PADW;
  Xs[(size_t)xc * NP + NCOL + p] = cmk(0.0f, 0.0f);
}

// --------------------------- Stage C: WMMA GEMM -----------------------------
typedef __attribute__((ext_vector_type(16))) __bf16 v16bf;
typedef __attribute__((ext_vector_type(8)))  float  v8f;

constexpr int TM = 64, TN = 32, TK = 32;

DEVFN v8f wmma_bf16(v16bf a, v16bf b, v8f c) {
  return __builtin_amdgcn_wmma_f32_16x16x32_bf16(false, a, false, b,
                                                 (short)0, c, false, false);
}

#if USE_TDM
typedef unsigned int uint32x4 __attribute__((ext_vector_type(4)));
typedef int          int32x4  __attribute__((ext_vector_type(4)));
typedef int          int32x8  __attribute__((ext_vector_type(8)));

// 2D tile load via Tensor Data Mover (D# groups per CDNA5 ISA).
// Elements are 8B (cplx). LDS rows padded by 2 cplx (pad_interval=64 DWORDs,
// pad_amount=4 DWORDs) so LDS rows stay 16B-aligned (stride 34 cplx = 272B).
DEVFN void tdm_load_2d(unsigned lds_off, const void* gptr,
                       int tile_w, int tile_h, long row_stride) {
  unsigned long long ga = (unsigned long long)(size_t)gptr;
  uint32x4 g0;
  g0[0] = 1u;                                          // count=1, user mode
  g0[1] = lds_off;                                     // lds_addr (bytes)
  g0[2] = (unsigned)(ga & 0xFFFFFFFFu);                // global_addr[31:0]
  g0[3] = (unsigned)((ga >> 32) & 0x01FFFFFFu) | (2u << 30);  // addr[56:32]|type=2
  const unsigned TD = 1u << 20;                        // large dims: no clipping
  int32x8 g1;
  g1[0] = (3 << 16) | (1 << 20) | (5 << 22) | (3 << 25); // 8B, pad_en, 64DW, +4DW
  g1[1] = (int)((TD & 0xFFFFu) << 16);                 // tensor_dim0 lo16
  g1[2] = (int)((TD >> 16) | ((TD & 0xFFFFu) << 16));  // dim0 hi | dim1 lo
  g1[3] = (int)((TD >> 16) | ((unsigned)tile_w << 16));// dim1 hi | tile_dim0
  g1[4] = tile_h;                                      // tile_dim1 (tile_dim2=0)
  g1[5] = (int)(unsigned)row_stride;                   // tensor_dim0_stride lo
  g1[6] = 0;
  g1[7] = 0;
  int32x4 z4 = {0, 0, 0, 0};
#if __clang_major__ >= 23
  int32x8 z8 = {0, 0, 0, 0, 0, 0, 0, 0};
  __builtin_amdgcn_tensor_load_to_lds(g0, g1, z4, z4, z8, 0);
#else
  __builtin_amdgcn_tensor_load_to_lds(g0, g1, z4, z4, 0);
#endif
}
#endif  // USE_TDM

__global__ __launch_bounds__(256)
void gemm_spectral(const cplx* __restrict__ Weff, const cplx* __restrict__ Xs,
                   cplx* __restrict__ Ys) {
  int x  = blockIdx.x;
  int mo = blockIdx.y * TM;
  int no = blockIdx.z * TN;

  // raw fp32 tiles, double-buffered; rows padded +2 cplx (matches TDM pad)
  __shared__ alignas(16) cplx Araw[2][TM][TK + 2];     // 2 x 17,408 B
  __shared__ alignas(16) cplx Xraw[2][TK][TN + 2];     // 2 x  8,704 B

  int tid  = threadIdx.x;
  int lane = tid & 31;
  int wv   = tid >> 5;
  int wm   = wv >> 1, wn = wv & 1;               // 4x2 subtiles of 16x16
  int hsel = lane >> 4, rr = lane & 15;

  v8f accRr = {}, accIi = {}, accRi = {}, accIr = {};

  const cplx* Wx = Weff + (size_t)x * CH * CH + (size_t)mo * CH;
  const cplx* Xx = Xs   + (size_t)x * CH * NP + no;

#if USE_TDM
  if (tid < 32) {                                // one wave drives the TDM
    tdm_load_2d((unsigned)(size_t)&Araw[0][0][0], Wx, TK, TM, CH);
    tdm_load_2d((unsigned)(size_t)&Xraw[0][0][0], Xx, TN, TK, NP);
  }
#endif

  for (int kb = 0, it = 0; kb < CH; kb += TK, ++it) {
    int cur = it & 1;
#if USE_TDM
    if (tid < 32) {
      if (kb + TK < CH) {                        // pipeline: issue next, then
        int nxt = cur ^ 1;                       // wait for current pair only
        tdm_load_2d((unsigned)(size_t)&Araw[nxt][0][0],
                    Wx + (kb + TK), TK, TM, CH);
        tdm_load_2d((unsigned)(size_t)&Xraw[nxt][0][0],
                    Xx + (size_t)(kb + TK) * NP, TN, TK, NP);
        __builtin_amdgcn_s_wait_tensorcnt(2);
      } else {
        __builtin_amdgcn_s_wait_tensorcnt(0);
      }
    }
#else
    for (int e = tid; e < TM * TK; e += 256)
      Araw[cur][e >> 5][e & 31] = Wx[(size_t)(e >> 5) * CH + kb + (e & 31)];
    for (int e = tid; e < TK * TN; e += 256)
      Xraw[cur][e >> 5][e & 31] = Xx[(size_t)(kb + (e >> 5)) * NP + (e & 31)];
    if (kb + TK < CH)                            // emits global_prefetch_b8
      __builtin_prefetch(Wx + (size_t)(tid & 63) * CH + kb + TK, 0, 1);
#endif
    __syncthreads();

    // ---- A fragments: lane reads two contiguous 64B runs of its row;
    //      all four bf16 variants (rH/rL/iH/iL) split in registers.
    //      Frag elem j <-> K = (j>>3)*16 + hsel*8 + (j&7).
    v16bf arh, arl, aih, ail;
    {
      const float4* q0 =
          reinterpret_cast<const float4*>(&Araw[cur][wm * 16 + rr][hsel * 8]);
      const float4* q1 =
          reinterpret_cast<const float4*>(&Araw[cur][wm * 16 + rr][16 + hsel * 8]);
#pragma unroll
      for (int t = 0; t < 8; ++t) {
        float4 f = (t < 4) ? q0[t] : q1[t - 4];  // 2 cplx per float4
        int j = 2 * t;
        __bf16 h;
        h = (__bf16)f.x; arh[j] = h;     arl[j] = (__bf16)(f.x - (float)h);
        h = (__bf16)f.y; aih[j] = h;     ail[j] = (__bf16)(f.y - (float)h);
        h = (__bf16)f.z; arh[j + 1] = h; arl[j + 1] = (__bf16)(f.z - (float)h);
        h = (__bf16)f.w; aih[j + 1] = h; ail[j + 1] = (__bf16)(f.w - (float)h);
      }
    }
    // ---- B fragments: frag elem e <-> K = (e>>3)*16 + hsel*8 + (e&7), column = lane&15.
    //      Row stride 272B => column walks banks with stride 4 (conflict-light).
    v16bf xrh, xrl, xih, xil;
    {
      int n = wn * 16 + (lane & 15);
#pragma unroll
      for (int e = 0; e < 16; ++e) {
        cplx v = Xraw[cur][(e >> 3) * 16 + hsel * 8 + (e & 7)][n];
        __bf16 h;
        h = (__bf16)v.x; xrh[e] = h; xrl[e] = (__bf16)(v.x - (float)h);
        h = (__bf16)v.y; xih[e] = h; xil[e] = (__bf16)(v.y - (float)h);
      }
    }

    // bf16x3 error-corrected complex MACs: 12 WMMAs / k-step
    accRr = wmma_bf16(arh, xrh, accRr);
    accRr = wmma_bf16(arh, xrl, accRr);
    accRr = wmma_bf16(arl, xrh, accRr);
    accIi = wmma_bf16(aih, xih, accIi);
    accIi = wmma_bf16(aih, xil, accIi);
    accIi = wmma_bf16(ail, xih, accIi);
    accRi = wmma_bf16(arh, xih, accRi);
    accRi = wmma_bf16(arh, xil, accRi);
    accRi = wmma_bf16(arl, xih, accRi);
    accIr = wmma_bf16(aih, xrh, accIr);
    accIr = wmma_bf16(aih, xrl, accIr);
    accIr = wmma_bf16(ail, xrh, accIr);
    __syncthreads();
  }

  // C/D layout: VGPR v -> M = v + 8*(lane>=16), N = lane&15
  cplx* Yx = Ys + (size_t)x * CH * NP;
  int n = lane & 15;
#pragma unroll
  for (int v = 0; v < 8; ++v) {
    int m = v + hsel * 8;
    Yx[(size_t)(mo + wm * 16 + m) * NP + no + wn * 16 + n] =
        cmk(accRr[v] - accIi[v], accRi[v] + accIr[v]);
  }
}

// ------------------------------ Stage D: col iFFT ---------------------------
__global__ __launch_bounds__(128) void ifft_cols(const cplx* __restrict__ Ys,
                                                 cplx* __restrict__ Yrow) {
  __shared__ cplx s[256];
  int idx = blockIdx.x;                          // (b*CH + o)*WC + y
  int y = idx % WC, bo = idx / WC;
  int o = bo % CH, b = bo / CH;
  int ncol = b * WC + y;
  const cplx* inp = Ys + (size_t)o * NP + ncol;
  int t = threadIdx.x;
  for (int k = t; k < 256; k += 128) s[brev8(k)] = inp[(size_t)k * CH * NP];
  __syncthreads();
  fft256(s, +1.0f);
  cplx* outp = Yrow + (size_t)bo * NLAT * WC + y;
  const float inv = 1.0f / 256.0f;
  for (int k = t; k < 256; k += 128)
    outp[(size_t)k * WC] = cmk(s[k].x * inv, s[k].y * inv);
}

// ------------------------------ Stage E: row irFFT --------------------------
__global__ __launch_bounds__(128) void irfft_rows(const cplx* __restrict__ Yrow,
                                                  float* __restrict__ out) {
  __shared__ cplx s[256];
  size_t row = blockIdx.x;                       // (b*CH + o)*NLAT + lat
  const cplx* X = Yrow + row * WC;
  int t = threadIdx.x;
  for (int k = t; k < 256; k += 128) {
    cplx Xk = X[k];
    cplx Xm = X[256 - k];
    cplx E = cmk(0.5f * (Xk.x + Xm.x), 0.5f * (Xk.y - Xm.y));
    cplx O = cmk(0.5f * (Xk.x - Xm.x), 0.5f * (Xk.y + Xm.y));
    float cs, sn;
    __sincosf(PI * (float)k / 256.0f, &cs, &sn);
    float p = O.x * cs - O.y * sn;
    float q = O.x * sn + O.y * cs;
    s[brev8(k)] = cmk(E.x - q, E.y + p);         // E + i(p+iq)
  }
  __syncthreads();
  fft256(s, +1.0f);
  float* op = out + row * NLON;
  const float inv = 1.0f / 256.0f;
  for (int n = t; n < 256; n += 128) {
    op[2 * n]     = s[n].x * inv;
    op[2 * n + 1] = s[n].y * inv;
  }
}

}  // namespace sc

// ---------------------------------------------------------------------------
extern "C" void kernel_launch(void* const* d_in, const int* in_sizes, int n_in,
                              void* d_out, int out_size, void* d_ws, size_t ws_size,
                              hipStream_t stream) {
  using namespace sc;
  (void)in_sizes; (void)n_in; (void)out_size; (void)ws_size;

  const float* x  = (const float*)d_in[0];
  const float* W  = (const float*)d_in[1];
  const float* lA = (const float*)d_in[2];
  const float* lB = (const float*)d_in[3];
  float* out = (float*)d_out;

  // workspace layout (cplx units)
  constexpr size_t BUF = (size_t)NLAT * CH * NP;           // 35,651,584 cplx
  cplx* ws0 = (cplx*)d_ws;        // Xrow -> later Ys
  cplx* ws1 = ws0 + BUF;          // Xs   -> later Yrow
  cplx* ws2 = ws1 + BUF;          // Weff (NLAT*CH*CH cplx)

  cplx* Xrow = ws0;
  cplx* Xs   = ws1;
  cplx* Weff = ws2;
  cplx* Ys   = ws0;
  cplx* Yrow = ws1;

  constexpr int ROWS = Bsz * CH * NLAT;     // 131072
  constexpr int COLS = Bsz * CH * WC;       // 131584
  constexpr int PADN = NLAT * CH * (NP - NCOL);

  sc::pad_zero<<<(PADN + 255) / 256, 256, 0, stream>>>(Xs);
  sc::fft_rows_fwd<<<ROWS, 128, 0, stream>>>(x, Xrow);
  sc::fft_cols_fwd<<<COLS, 128, 0, stream>>>(Xrow, Xs);
  sc::weff_kernel<<<NLAT, 256, 0, stream>>>(W, lA, lB, Weff);
  sc::gemm_spectral<<<dim3(NLAT, CH / TM, NP / TN), 256, 0, stream>>>(Weff, Xs, Ys);
  sc::ifft_cols<<<COLS, 128, 0, stream>>>(Ys, Yrow);
  sc::irfft_rows<<<ROWS, 128, 0, stream>>>(Yrow, out);

  // residual = x (second output of the tuple)
  hipMemcpyAsync(out + (size_t)Bsz * CH * NLAT * NLON, x,
                 (size_t)Bsz * CH * NLAT * NLON * sizeof(float),
                 hipMemcpyDeviceToDevice, stream);
}
